// RegionEncoder_48198122996214
// MI455X (gfx1250) — matrix operand order repaired and checked by reading, not compiled, not run on hardware
//
#include <hip/hip_runtime.h>
#include <hip/hip_bf16.h>

// ---------------------------------------------------------------------------
// RegionEncoder (2x TransformerConv + LN + GELU) for MI455X / gfx1250.
//   - Dense projections via v_wmma_f32_16x16x32_bf16 (fused Q|K|V|S GEMM)
//   - Edge softmax/aggregation via float atomics (L2-resident: 192MB >> 128MB
//     working set)
//   - LayerNorm + exact GELU fused finalize, wave32 shuffle reductions
// ---------------------------------------------------------------------------

#define NN 50000
#define EE 800000
#define DM 128
#define HH 4
#define DH 32

typedef __attribute__((ext_vector_type(8)))  __bf16 v8bf;
typedef __attribute__((ext_vector_type(16))) __bf16 v16bf;
typedef __attribute__((ext_vector_type(8)))  float  v8f;

// ------------------------- ordered-float <-> uint ---------------------------
__device__ __forceinline__ unsigned ordF(float f) {
    unsigned u = __float_as_uint(f);
    return (u >> 31) ? ~u : (u ^ 0x80000000u);
}
__device__ __forceinline__ float unordF(unsigned u) {
    u = (u >> 31) ? (u ^ 0x80000000u) : ~u;
    return __uint_as_float(u);
}

// ----------------------------- zero fill ------------------------------------
__global__ void zero_f32(float* p, long long n) {
    long long i = (long long)blockIdx.x * blockDim.x + threadIdx.x;
    long long stride = (long long)gridDim.x * blockDim.x;
    for (; i < n; i += stride) p[i] = 0.0f;
}

// ----------------------------- fp32 -> bf16 ---------------------------------
__global__ void conv_bf16(const float* __restrict__ x, __bf16* __restrict__ y,
                          long long n) {
    long long i = (long long)blockIdx.x * blockDim.x + threadIdx.x;
    long long stride = (long long)gridDim.x * blockDim.x;
    for (; i < n; i += stride) y[i] = (__bf16)x[i];
}

// ----------------- weight prep: transpose + concat + bf16 -------------------
// Wt[col][k] (512x128, col-major vs original) so B-fragments are contiguous.
__global__ void prep_weights(const float* __restrict__ wq, const float* __restrict__ wk,
                             const float* __restrict__ wv, const float* __restrict__ ws,
                             const float* __restrict__ bq, const float* __restrict__ bk,
                             const float* __restrict__ bv, const float* __restrict__ bs,
                             __bf16* __restrict__ Wt, float* __restrict__ biascat) {
    int idx = blockIdx.x * blockDim.x + threadIdx.x;   // 512*128
    if (idx >= 512 * 128) return;
    int col = idx >> 7;       // output column 0..511
    int k   = idx & 127;      // reduction index
    int which = col >> 7;     // 0=Q 1=K 2=V 3=S
    int n = col & 127;
    const float* W = (which == 0) ? wq : (which == 1) ? wk : (which == 2) ? wv : ws;
    Wt[col * 128 + k] = (__bf16)W[k * 128 + n];
    if (k == 0) {
        const float* B = (which == 0) ? bq : (which == 1) ? bk : (which == 2) ? bv : bs;
        biascat[col] = B[n];
    }
}

// ------------------------- WMMA fragment loaders ----------------------------
// A: 16x32 bf16, ISA 7.12.2: lanes 0-15 row M=lane, VGPR0-3 => K=kb+0..7,
// VGPR4-7 => K=kb+16..23; lanes 16-31 same rows, K ranges +8.
__device__ __forceinline__ v16bf load_frag_a(const __bf16* __restrict__ X,
                                             int row0, int kb, int lane) {
    int half = lane >> 4;
    int m = row0 + (lane & 15);
    const __bf16* p = X + (long long)m * DM + kb + half * 8;
    v8bf c0 = *(const v8bf*)(p);
    v8bf c1 = *(const v8bf*)(p + 16);
    v16bf a;
#pragma unroll
    for (int i = 0; i < 8; ++i) { a[i] = c0[i]; a[i + 8] = c1[i]; }
    return a;
}

// B: 32x16 bf16 from transposed weights Wt[n][k]: lanes 0-15 col N=lane,
// K=kb+0..7,16..23; lanes 16-31 same cols, K ranges +8.
__device__ __forceinline__ v16bf load_frag_b(const __bf16* __restrict__ Wt,
                                             int col0, int kb, int lane) {
    int half = lane >> 4;
    int n = col0 + (lane & 15);
    const __bf16* p = Wt + (long long)n * DM + kb + half * 8;
    v8bf c0 = *(const v8bf*)(p);
    v8bf c1 = *(const v8bf*)(p + 16);
    v16bf b;
#pragma unroll
    for (int i = 0; i < 8; ++i) { b[i] = c0[i]; b[i + 8] = c1[i]; }
    return b;
}

// ------------------ fused projection GEMM: X @ [Wq|Wk|Wv|Ws] ----------------
// 256 threads = 8 waves. Block owns a 16-row tile of X; wave w owns output
// columns [w*64, w*64+64). K-loop: 4 x v_wmma_f32_16x16x32_bf16 per tile.
__global__ void proj_gemm(const __bf16* __restrict__ Xbf, const __bf16* __restrict__ Wt,
                          const float* __restrict__ biascat, float* __restrict__ QKVS) {
    int wave = threadIdx.x >> 5;
    int lane = threadIdx.x & 31;
    int row0 = blockIdx.x * 16;
    int col0 = wave * 64;

    v8f acc0 = {}, acc1 = {}, acc2 = {}, acc3 = {};
#pragma unroll
    for (int kb = 0; kb < DM; kb += 32) {
        v16bf a = load_frag_a(Xbf, row0, kb, lane);
        v16bf b0 = load_frag_b(Wt, col0 + 0,  kb, lane);
        acc0 = __builtin_amdgcn_wmma_f32_16x16x32_bf16(false, a, false, b0, (short)0, acc0, false, false);
        v16bf b1 = load_frag_b(Wt, col0 + 16, kb, lane);
        acc1 = __builtin_amdgcn_wmma_f32_16x16x32_bf16(false, a, false, b1, (short)0, acc1, false, false);
        v16bf b2 = load_frag_b(Wt, col0 + 32, kb, lane);
        acc2 = __builtin_amdgcn_wmma_f32_16x16x32_bf16(false, a, false, b2, (short)0, acc2, false, false);
        v16bf b3 = load_frag_b(Wt, col0 + 48, kb, lane);
        acc3 = __builtin_amdgcn_wmma_f32_16x16x32_bf16(false, a, false, b3, (short)0, acc3, false, false);
    }

    // C/D layout: VGPR r -> row M = r (lanes 0-15) / 8+r (lanes 16-31); N = lane&15
    int n  = lane & 15;
    int mh = (lane >> 4) * 8;
#pragma unroll
    for (int r = 0; r < 8; ++r) {
        long long m = row0 + mh + r;
        int c0 = col0 + n;
        QKVS[m * 512 + c0 +  0] = acc0[r] + biascat[c0 +  0];
        QKVS[m * 512 + c0 + 16] = acc1[r] + biascat[c0 + 16];
        QKVS[m * 512 + c0 + 32] = acc2[r] + biascat[c0 + 32];
        QKVS[m * 512 + c0 + 48] = acc3[r] + biascat[c0 + 48];
    }
}

// ---------------- edge pass 1: attention logits + segment max ---------------
__global__ void edge_alpha(const long long* __restrict__ ei, const float* __restrict__ ew,
                           const float* __restrict__ we, const float* __restrict__ QKVS,
                           float* __restrict__ alpha, unsigned* __restrict__ amaxU) {
    long long t = (long long)blockIdx.x * blockDim.x + threadIdx.x;
    if (t >= (long long)EE * HH) return;
    long long e = t >> 2;
    int h = (int)(t & 3);
    int src = (int)ei[e];
    int dst = (int)ei[EE + e];
    float w = ew[e];
    const float* q  = QKVS + (long long)dst * 512 + h * DH;          // Q block
    const float* k  = QKVS + (long long)src * 512 + 128 + h * DH;    // K block
    const float* ev = we + h * DH;
    float acc = 0.0f;
#pragma unroll
    for (int j = 0; j < DH; j += 4) {
        float4 qv = *(const float4*)(q + j);
        float4 kv = *(const float4*)(k + j);
        float4 e4 = *(const float4*)(ev + j);
        acc += qv.x * (kv.x + w * e4.x);
        acc += qv.y * (kv.y + w * e4.y);
        acc += qv.z * (kv.z + w * e4.z);
        acc += qv.w * (kv.w + w * e4.w);
    }
    float a = acc * 0.17677669529663687f;   // 1/sqrt(32)
    alpha[e * HH + h] = a;
    atomicMax(&amaxU[(long long)dst * HH + h], ordF(a));
}

// ---------------- edge pass 2: exp(alpha - max) + segment sum ---------------
__global__ void edge_exp(const long long* __restrict__ ei, float* __restrict__ alpha,
                         const unsigned* __restrict__ amaxU, float* __restrict__ denom) {
    long long t = (long long)blockIdx.x * blockDim.x + threadIdx.x;
    if (t >= (long long)EE * HH) return;
    long long e = t >> 2;
    int h = (int)(t & 3);
    int dst = (int)ei[EE + e];
    float m  = unordF(amaxU[(long long)dst * HH + h]);
    float ex = __expf(alpha[e * HH + h] - m);
    alpha[e * HH + h] = ex;
    atomicAdd(&denom[(long long)dst * HH + h], ex);
}

// ---------------- edge pass 3: weighted message scatter-add -----------------
__global__ void edge_msg(const long long* __restrict__ ei, const float* __restrict__ ew,
                         const float* __restrict__ we, const float* __restrict__ QKVS,
                         const float* __restrict__ alpha, const float* __restrict__ denom,
                         float* __restrict__ agg) {
    long long t = (long long)blockIdx.x * blockDim.x + threadIdx.x;
    if (t >= (long long)EE * DM) return;
    long long e = t >> 7;
    int f = (int)(t & 127);
    int h = f >> 5;
    int src = (int)ei[e];
    int dst = (int)ei[EE + e];
    float attn = alpha[e * HH + h] / denom[(long long)dst * HH + h];
    float v = QKVS[(long long)src * 512 + 256 + f] + ew[e] * we[f];  // V block + edge
    atomicAdd(&agg[(long long)dst * DM + f], v * attn);
}

// ------------- finalize: +skip, LayerNorm, exact GELU, emit bf16 ------------
__global__ void finalize_ln_gelu(const float* __restrict__ agg, const float* __restrict__ QKVS,
                                 const float* __restrict__ g, const float* __restrict__ b,
                                 float* __restrict__ outF, __bf16* __restrict__ outB) {
    int wave = threadIdx.x >> 5;
    int lane = threadIdx.x & 31;
    int node = blockIdx.x * 8 + wave;
    if (node >= NN) return;
    float v[4];
#pragma unroll
    for (int j = 0; j < 4; ++j) {
        int f = lane + 32 * j;
        v[j] = agg[(long long)node * DM + f] + QKVS[(long long)node * 512 + 384 + f];
    }
    float s = v[0] + v[1] + v[2] + v[3];
#pragma unroll
    for (int off = 16; off > 0; off >>= 1) s += __shfl_xor(s, off);
    float mean = s * (1.0f / 128.0f);
    float d[4], ss = 0.0f;
#pragma unroll
    for (int j = 0; j < 4; ++j) { d[j] = v[j] - mean; ss += d[j] * d[j]; }
#pragma unroll
    for (int off = 16; off > 0; off >>= 1) ss += __shfl_xor(ss, off);
    float rstd = rsqrtf(ss * (1.0f / 128.0f) + 1e-5f);
#pragma unroll
    for (int j = 0; j < 4; ++j) {
        int f = lane + 32 * j;
        float y  = d[j] * rstd * g[f] + b[f];
        float ge = 0.5f * y * (1.0f + erff(y * 0.70710678118654752f));
        if (outF) outF[(long long)node * DM + f] = ge;
        if (outB) outB[(long long)node * DM + f] = (__bf16)ge;
    }
}

// ---------------------------------------------------------------------------
static inline size_t alignup(size_t x) { return (x + 255) & ~(size_t)255; }

extern "C" void kernel_launch(void* const* d_in, const int* in_sizes, int n_in,
                              void* d_out, int out_size, void* d_ws, size_t ws_size,
                              hipStream_t stream) {
    (void)in_sizes; (void)n_in; (void)out_size; (void)ws_size;

    const float*     x  = (const float*)d_in[0];
    const long long* ei = (const long long*)d_in[1];
    const float*     ew = (const float*)d_in[2];

    // workspace carve-up (amax/denom/agg adjacent for one-shot zeroing)
    char* w = (char*)d_ws;
    __bf16* Xbf    = (__bf16*)w;                 w += alignup((size_t)NN * DM * 2);
    __bf16* Wt     = (__bf16*)w;                 w += alignup((size_t)512 * DM * 2);
    float*  biascat= (float*)w;                  w += alignup((size_t)512 * 4);
    float*  QKVS   = (float*)w;                  w += alignup((size_t)NN * 512 * 4);
    unsigned* amaxU= (unsigned*)w;               w += (size_t)NN * HH * 4;       // 800000B, 256-mult
    float*  denom  = (float*)w;                  w += (size_t)NN * HH * 4;
    float*  agg    = (float*)w;                  w += alignup((size_t)NN * DM * 4);
    float*  alpha  = (float*)w;                  w += alignup((size_t)EE * HH * 4);

    const int T = 256;
    // initial bf16 copy of node features
    conv_bf16<<<4096, T, 0, stream>>>(x, Xbf, (long long)NN * DM);

    for (int layer = 0; layer < 2; ++layer) {
        const int p = 3 + layer * 11;
        const float* wq = (const float*)d_in[p + 0];
        const float* bq = (const float*)d_in[p + 1];
        const float* wk = (const float*)d_in[p + 2];
        const float* bk = (const float*)d_in[p + 3];
        const float* wv = (const float*)d_in[p + 4];
        const float* bv = (const float*)d_in[p + 5];
        const float* we = (const float*)d_in[p + 6];
        const float* ws = (const float*)d_in[p + 7];
        const float* bs = (const float*)d_in[p + 8];
        const float* lg = (const float*)d_in[p + 9];
        const float* lb = (const float*)d_in[p + 10];

        prep_weights<<<(512 * 128 + T - 1) / T, T, 0, stream>>>(
            wq, wk, wv, ws, bq, bk, bv, bs, Wt, biascat);

        proj_gemm<<<NN / 16, T, 0, stream>>>(Xbf, Wt, biascat, QKVS);

        // zero amax | denom | agg in one pass (contiguous region)
        zero_f32<<<4096, T, 0, stream>>>((float*)amaxU,
                                         (long long)NN * (HH + HH + DM));

        long long eh = (long long)EE * HH;
        edge_alpha<<<(unsigned)((eh + T - 1) / T), T, 0, stream>>>(ei, ew, we, QKVS, alpha, amaxU);
        edge_exp  <<<(unsigned)((eh + T - 1) / T), T, 0, stream>>>(ei, alpha, amaxU, denom);
        long long ef = (long long)EE * DM;
        edge_msg  <<<(unsigned)((ef + T - 1) / T), T, 0, stream>>>(ei, ew, we, QKVS, alpha, denom, agg);

        float*  outF = (layer == 1) ? (float*)d_out : nullptr;
        __bf16* outB = (layer == 0) ? Xbf : nullptr;
        finalize_ln_gelu<<<(NN + 7) / 8, T, 0, stream>>>(agg, QKVS, lg, lb, outF, outB);
    }
}
